// mBP_model_14920716387122
// MI455X (gfx1250) — matrix operand order repaired and checked by reading, not compiled, not run on hardware
//
#include <hip/hip_runtime.h>
#include <hip/hip_fp16.h>

// ---------------- problem constants (match reference) ----------------
#define NAT    192
#define NPAIR  (NAT * NAT)
#define DPAD   128      // padded descriptor width (feat = 96)
#define HID    128
#define PI_F   3.14159265358979f

typedef __attribute__((ext_vector_type(16))) _Float16 v16h;
typedef __attribute__((ext_vector_type(8)))  float    v8f;

// power-series approximation of exp(-x): (1/(1+x/64+1e-8))^64
__device__ __forceinline__ float app_gaussian(float x) {
    float a  = 1.0f / (1.0f + x * (1.0f / 64.0f) + 1e-8f);
    float a2 = a * a, a4 = a2 * a2, a8 = a4 * a4, a16 = a8 * a8, a32 = a16 * a16;
    return a32 * a32;
}
__device__ __forceinline__ float pow8(float b) {
    float b2 = b * b, b4 = b2 * b2;
    return b4 * b4;
}

// ---------------- kernel 1: pairwise geometry ----------------
__global__ __launch_bounds__(256)
void pair_kernel(const float* __restrict__ pos, const float* __restrict__ spc,
                 float* __restrict__ rn, float* __restrict__ ux, float* __restrict__ uy,
                 float* __restrict__ uz, float* __restrict__ pw, float* __restrict__ sfc) {
    int p = blockIdx.x * blockDim.x + threadIdx.x;
    if (p >= NPAIR) return;
    int i = p / NAT, j = p % NAT;
    float dx = pos[3*i+0] - pos[3*j+0];
    float dy = pos[3*i+1] - pos[3*j+1];
    float dz = pos[3*i+2] - pos[3*j+2];
    float r   = sqrtf(dx*dx + dy*dy + dz*dz + 1e-12f);
    float inv = 1.0f / (r + 1e-20f);
    rn[p] = r;
    ux[p] = dx * inv; uy[p] = dy * inv; uz[p] = dz * inv;
    float fa = (r <= 4.0f) ? 0.5f * (1.0f + cosf(PI_F * r * 0.25f))        : 0.0f;
    float fr = (r <= 6.0f) ? 0.5f * (1.0f + cosf(PI_F * r * (1.0f/6.0f))) : 0.0f;
    float s = spc[j];
    pw[p]  = s * fa;
    sfc[p] = s * fr;
}

// ---------------- kernel 2: radial descriptors ----------------
// one wave per atom, lane = radial shell index (32 shells == 32 lanes)
__global__ __launch_bounds__(32)
void radial_kernel(const float* __restrict__ rn, const float* __restrict__ sfc,
                   float* __restrict__ desc) {
    int i = blockIdx.x;
    int s = threadIdx.x;                       // 0..31
    float Rs  = 0.25f + (float)s * ((6.0f - 0.25f) / 32.0f);
    float acc = 0.0f;
    for (int j = 0; j < NAT; ++j) {
        float d = rn[i*NAT + j] - Rs;
        acc += sfc[i*NAT + j] * app_gaussian(0.5f * d * d);   // WIDTH = 0.5
    }
    desc[i*DPAD + s] = acc;
}

// ---------------- kernel 3: angular (triplet) descriptors ----------------
// one block per atom i; per-thread 8x8 register accumulators; exploits exact
// (j,k) symmetry: rows paired (t, 191-t) for load balance, weight 2 off-diagonal.
__global__ __launch_bounds__(96)
void angular_kernel(const float* __restrict__ rn, const float* __restrict__ ux,
                    const float* __restrict__ uy, const float* __restrict__ uz,
                    const float* __restrict__ pw, float* __restrict__ desc) {
    __shared__ float s_rn[NAT], s_ux[NAT], s_uy[NAT], s_uz[NAT], s_pw[NAT];
    __shared__ float s_G[64];

    const int i = blockIdx.x;
    const int t = threadIdx.x;                 // 0..95

    for (int j = t; j < NAT; j += 96) {
        int p = i*NAT + j;
        s_rn[j] = rn[p]; s_ux[j] = ux[p]; s_uy[j] = uy[p]; s_uz[j] = uz[p];
        s_pw[j] = pw[p];
    }
    if (t < 64) s_G[t] = 0.0f;
    __syncthreads();

    // cos/sin of theta_s = t*pi/8
    const float ct[8] = { 1.0f,  0.92387953f,  0.70710678f,  0.38268343f,
                          0.0f, -0.38268343f, -0.70710678f, -0.92387953f };
    const float st[8] = { 0.0f,  0.38268343f,  0.70710678f,  0.92387953f,
                          1.0f,  0.92387953f,  0.70710678f,  0.38268343f };

    float G[8][8];
    #pragma unroll
    for (int a = 0; a < 8; ++a) {
        #pragma unroll
        for (int b = 0; b < 8; ++b) G[a][b] = 0.0f;
    }

    const int rows[2] = { t, NAT - 1 - t };    // balanced: (192-t) + (t+1) = 193 iters
    for (int rsel = 0; rsel < 2; ++rsel) {
        const int j = rows[rsel];
        const float pj = s_pw[j];
        if (pj == 0.0f) continue;              // beyond rc_ang -> zero contribution
        const float rj  = s_rn[j];
        const float uxj = s_ux[j], uyj = s_uy[j], uzj = s_uz[j];
        for (int k = j; k < NAT; ++k) {
            float pv = pj * s_pw[k];
            if (pv == 0.0f) continue;
            pv *= (k == j) ? 1.0f : 2.0f;      // symmetric (j,k)<->(k,j)
            float c  = uxj * s_ux[k] + uyj * s_uy[k] + uzj * s_uz[k];
            float sn = sqrtf(fmaxf(1.0f - c * c, 0.0f) + 1e-6f);
            float rr = 0.5f * (rj + s_rn[k]);

            float angv[8], radg[8];
            #pragma unroll
            for (int tt = 0; tt < 8; ++tt)
                angv[tt] = pow8((1.0f + c * ct[tt] + sn * st[tt]) * 0.5f); // zeta=8
            #pragma unroll
            for (int ss = 0; ss < 8; ++ss) {
                float d = rr - (0.25f + (float)ss * 0.46875f);  // Rs_ang
                radg[ss] = app_gaussian(0.5f * d * d);          // WIDTH_ANG = 0.5
            }
            #pragma unroll
            for (int ss = 0; ss < 8; ++ss) {
                float pr = pv * radg[ss];
                #pragma unroll
                for (int tt = 0; tt < 8; ++tt)
                    G[ss][tt] += pr * angv[tt];
            }
        }
    }

    #pragma unroll
    for (int ss = 0; ss < 8; ++ss) {
        #pragma unroll
        for (int tt = 0; tt < 8; ++tt)
            atomicAdd(&s_G[ss*8 + tt], G[ss][tt]);   // ds_add_f32
    }
    __syncthreads();

    if (t < 64)  desc[i*DPAD + 32 + t]        = s_G[t];   // cols 32..95
    if (t >= 64) desc[i*DPAD + 96 + (t - 64)] = 0.0f;     // zero pad cols 96..127
}

// ---------------- kernel 4/5: WMMA GEMM + bias + tanh ----------------
// Out[192,128] = tanh(A[192,K (lda=128)] @ W[K,128] + bias).
// One wave computes a 16x32 output tile (two 16x16 WMMA accumulators) so the
// A fragment (16 scattered f32->f16 converts) is reused across two WMMAs.
// K is a template parameter -> fully unrolled K loop, static wmma chain.
template <int K>
__global__ __launch_bounds__(32)
void gemm_tanh_kernel(const float* __restrict__ A, const float* __restrict__ W,
                      const float* __restrict__ bias, float* __restrict__ Out) {
    const int tileM = blockIdx.x * 16;
    const int tileN = blockIdx.y * 32;         // two adjacent 16-wide N tiles
    const int lane  = threadIdx.x;             // wave32, EXEC all ones
    const int g     = lane >> 4;               // K half-group
    const int ma    = lane & 15;               // A row within tile

    v8f acc0 = {};
    v8f acc1 = {};
    #pragma unroll
    for (int k0 = 0; k0 < K; k0 += 32) {
        // A fragment: 16x32 f16, CDNA5 layout (lanes 0-15: K 0-7/16-23, lanes 16-31: +8)
        v16h a, b0, b1;
        const float* arow = A + (tileM + ma) * DPAD + k0;
        #pragma unroll
        for (int h = 0; h < 16; ++h) {
            int vv = h >> 1, lo = h & 1;
            int kk = (vv < 4) ? (8*g + 2*vv + lo) : (16 + 8*g + 2*(vv - 4) + lo);
            a[h] = (_Float16)arow[kk];
        }
        // B fragments: 32x16 f16, lane%16 = N column, K in same order as A (lanes 0-15: K 0-7/16-23, lanes 16-31: +8)
        const float* wrow = W + k0 * HID + tileN + ma;
        #pragma unroll
        for (int h = 0; h < 16; ++h) {
            int vv = h >> 1, lo = h & 1;
            int kk = (vv < 4) ? (8*g + 2*vv + lo) : (16 + 8*g + 2*(vv - 4) + lo);
            b0[h] = (_Float16)wrow[kk * HID];
            b1[h] = (_Float16)wrow[kk * HID + 16];
        }

        if (k0 + 32 < K)
            __builtin_prefetch(W + (k0 + 32 + lane) * HID + tileN, 0, 0); // global_prefetch_b8

        acc0 = __builtin_amdgcn_wmma_f32_16x16x32_f16(
            false, a, false, b0, (short)0, acc0, false, false);
        acc1 = __builtin_amdgcn_wmma_f32_16x16x32_f16(
            false, a, false, b1, (short)0, acc1, false, false);
    }

    // D layout: lane L, VGPR v -> M = 8*(L/16)+v, N = L%16
    const int n = lane & 15;
    #pragma unroll
    for (int vv = 0; vv < 8; ++vv) {
        int m = 8*g + vv;
        float v0 = acc0[vv] + bias[tileN + n];
        float v1 = acc1[vv] + bias[tileN + 16 + n];
        Out[(tileM + m) * HID + tileN + n]      = tanhf(v0);
        Out[(tileM + m) * HID + tileN + 16 + n] = tanhf(v1);
    }
}

// ---------------- kernel 6: final layer + total energy ----------------
__global__ __launch_bounds__(192)
void energy_kernel(const float* __restrict__ H2, const float* __restrict__ W3,
                   const float* __restrict__ b3, float* __restrict__ out) {
    __shared__ float part[NAT];
    int a = threadIdx.x;                       // one atom per thread
    float e = 0.0f;
    #pragma unroll 4
    for (int h = 0; h < HID; ++h) e += H2[a*HID + h] * W3[h];
    part[a] = e + b3[0];
    __syncthreads();
    if (a == 0) {
        float s = 0.0f;
        for (int q = 0; q < NAT; ++q) s += part[q];
        out[0] = s;
    }
}

// ---------------- host launcher ----------------
extern "C" void kernel_launch(void* const* d_in, const int* in_sizes, int n_in,
                              void* d_out, int out_size, void* d_ws, size_t ws_size,
                              hipStream_t stream) {
    const float* pos = (const float*)d_in[0];   // [192,3]
    const float* spc = (const float*)d_in[1];   // [192]
    const float* W1  = (const float*)d_in[2];   // [96,128]
    const float* b1  = (const float*)d_in[3];   // [128]
    const float* W2  = (const float*)d_in[4];   // [128,128]
    const float* b2  = (const float*)d_in[5];   // [128]
    const float* W3  = (const float*)d_in[6];   // [128,1]
    const float* b3  = (const float*)d_in[7];   // [1]
    float* out = (float*)d_out;

    float* ws   = (float*)d_ws;                 // ~1.2 MB total, fits easily in L2
    float* rn   = ws + 0 * NPAIR;
    float* ux   = ws + 1 * NPAIR;
    float* uy   = ws + 2 * NPAIR;
    float* uz   = ws + 3 * NPAIR;
    float* pw   = ws + 4 * NPAIR;
    float* sfc  = ws + 5 * NPAIR;
    float* desc = ws + 6 * NPAIR;               // [192,128] padded descriptors
    float* H1   = desc + NAT * DPAD;            // [192,128]
    float* H2   = H1   + NAT * HID;             // [192,128]

    pair_kernel<<<(NPAIR + 255) / 256, 256, 0, stream>>>(pos, spc, rn, ux, uy, uz, pw, sfc);
    radial_kernel<<<NAT, 32, 0, stream>>>(rn, sfc, desc);
    angular_kernel<<<NAT, 96, 0, stream>>>(rn, ux, uy, uz, pw, desc);
    gemm_tanh_kernel<96><<<dim3(NAT/16, HID/32), 32, 0, stream>>>(desc, W1, b1, H1);
    gemm_tanh_kernel<128><<<dim3(NAT/16, HID/32), 32, 0, stream>>>(H1, W2, b2, H2);
    energy_kernel<<<1, NAT, 0, stream>>>(H2, W3, b3, out);
}
